// GNN_topexpert_32263794327783
// MI455X (gfx1250) — matrix operand order repaired and checked by reading, not compiled, not run on hardware
//
#include <hip/hip_runtime.h>
#include <hip/hip_bf16.h>

#define EMB     300
#define HID     600
#define KP1     320     // EMB padded to multiple of 32
#define KP2     608     // HID padded to multiple of 32
#define NNODES  50000
#define NEDGES  256000
#define NLAYER  5
#define EPS_BN  1e-5f

typedef __attribute__((ext_vector_type(8)))  float    v8f;
typedef __attribute__((ext_vector_type(8)))  _Float16 v8h;
typedef __attribute__((ext_vector_type(16))) _Float16 v16h;

__device__ __forceinline__ void atomic_add_f32(float* p, float v) {
    (void)__hip_atomic_fetch_add(p, v, __ATOMIC_RELAXED, __HIP_MEMORY_SCOPE_AGENT);
}

// ---------------------------------------------------------------------------
// h0 = x_emb1[x[:,0]] + x_emb2[x[:,1]]   (f32)
// ---------------------------------------------------------------------------
__global__ void k_node_emb(const int* __restrict__ x,
                           const float* __restrict__ e1,
                           const float* __restrict__ e2,
                           float* __restrict__ h) {
    int n = blockIdx.x;
    int f = threadIdx.x;
    if (f >= EMB) return;
    int x0 = x[n * 2 + 0];
    int x1 = x[n * 2 + 1];
    h[(size_t)n * EMB + f] = e1[(size_t)x0 * EMB + f] + e2[(size_t)x1 * EMB + f];
}

__global__ void k_zero(float* __restrict__ p, int n) {
    int i = blockIdx.x * blockDim.x + threadIdx.x;
    if (i < n) p[i] = 0.0f;
}

// ---------------------------------------------------------------------------
// Scatter: msg = h[row] + edge_emb1[ea0] + edge_emb2[ea1]; agg[col] += msg
// One block per edge (self-loops appended); 300 active lanes over features.
// ---------------------------------------------------------------------------
__global__ void k_scatter(const float* __restrict__ h,
                          const int* __restrict__ edge_index,
                          const int* __restrict__ edge_attr,
                          const float* __restrict__ ee1,   // [6,EMB] (layer slice)
                          const float* __restrict__ ee2,   // [3,EMB] (layer slice)
                          float* __restrict__ agg) {
    int e = blockIdx.x;
    int f = threadIdx.x;
    if (f >= EMB) return;
    int row, col, a0, a1;
    if (e < NEDGES) {
        row = edge_index[e];
        col = edge_index[NEDGES + e];
        a0  = edge_attr[e * 2 + 0];
        a1  = edge_attr[e * 2 + 1];
    } else {
        int n = e - NEDGES;   // self loop
        row = n; col = n; a0 = 4; a1 = 0;
    }
    float m = h[(size_t)row * EMB + f] + ee1[(size_t)a0 * EMB + f] + ee2[(size_t)a1 * EMB + f];
    atomic_add_f32(&agg[(size_t)col * EMB + f], m);
}

// ---------------------------------------------------------------------------
// Convert agg f32 [N,EMB] -> f16 [N,KP1], zero-padding columns EMB..KP1-1
// ---------------------------------------------------------------------------
__global__ void k_cvt_agg(const float* __restrict__ agg, _Float16* __restrict__ aggH) {
    int n = blockIdx.x;
    int f = threadIdx.x;          // blockDim = KP1 = 320
    float v = (f < EMB) ? agg[(size_t)n * EMB + f] : 0.0f;
    aggH[(size_t)n * KP1 + f] = (_Float16)v;
}

// ---------------------------------------------------------------------------
// Transpose + convert weights: W f32 [K,N] -> Wt f16 [N,Kp], zero pad K..Kp-1
// ---------------------------------------------------------------------------
__global__ void k_cvt_w(const float* __restrict__ W, _Float16* __restrict__ Wt,
                        int K, int N, int Kp) {
    int n = blockIdx.x;           // output row (weight column)
    int k = threadIdx.x;          // blockDim = Kp (multiple of 32)
    if (k >= Kp) return;
    float v = (k < K) ? W[(size_t)k * N + n] : 0.0f;
    Wt[(size_t)n * Kp + k] = (_Float16)v;
}

// ---------------------------------------------------------------------------
// F16 WMMA GEMM (f32 accumulate): C[M,N] = act(A[M,Kp] @ Bt[N,Kp]^T + bias[N])
// One wave computes a 16x32 output slab (two 16x16x32 WMMA tiles sharing A).
// KP is a template constant -> fully unrolled, immediate-offset b128 loads.
//
// V_WMMA_F32_16X16X32_F16 k order used by this code (wave32; same for A and B):
//   A 16x32 : lane row M=lane&15; halves j: K = k0 + j + (hi?16:0)
//             => contiguous 16 halves in the A row
//   B 32x16 : lane col N=lane&15; halves j: K = k0 + j + (hi?16:0)
//             => contiguous 16 halves in the transposed-weight row
//   C/D     : VGPR r -> (M = m0 + r + (hi?8:0), N = n0 + (lane&15))
// ---------------------------------------------------------------------------
template <int KP, bool OUT16>
__global__ void k_gemm_wmma_f16(const _Float16* __restrict__ A,   // [M, KP]
                                const _Float16* __restrict__ Bt,  // [N, KP]
                                const float*    __restrict__ bias,// [N]
                                void* __restrict__ Cout,
                                int N, int Nstore, int Cstride,
                                int tilesM, int relu) {
    int wave    = (blockIdx.x * blockDim.x + threadIdx.x) >> 5;
    int lane    = threadIdx.x & 31;
    int tilesN2 = (N + 31) >> 5;
    if (wave >= tilesM * tilesN2) return;   // wave-uniform exit (EXEC all-1s)

    int tm = wave % tilesM;
    int tn = wave / tilesM;
    int m0 = tm << 4;
    int n0 = tn << 5;
    int lm = lane & 15;
    bool hi = lane >= 16;

    int nc0 = n0 + lm;
    int nc1 = n0 + 16 + lm;
    int cc0 = min(nc0, N - 1);              // clamped, always-valid B rows;
    int cc1 = min(nc1, N - 1);              // garbage cols are masked at store

    const _Float16* ap  = A  + (size_t)(m0 + lm) * KP + (hi ? 16 : 0);
    const _Float16* bp0 = Bt + (size_t)cc0 * KP + (hi ? 16 : 0);
    const _Float16* bp1 = Bt + (size_t)cc1 * KP + (hi ? 16 : 0);

    v8f acc0 = {};
    v8f acc1 = {};
    union Frag { v16h v; v8h h[2]; };

    #pragma unroll
    for (int k0 = 0; k0 < KP; k0 += 32) {
        Frag a, b0, b1;
        const v8h* pa  = (const v8h*)(ap  + k0);
        const v8h* pb0 = (const v8h*)(bp0 + k0);
        const v8h* pb1 = (const v8h*)(bp1 + k0);
        a.h[0]  = pa[0];   // K = k0+off .. +7
        a.h[1]  = pa[1];   // K = k0+8+off .. +7
        b0.h[0] = pb0[0];
        b0.h[1] = pb0[1];
        b1.h[0] = pb1[0];
        b1.h[1] = pb1[1];
        acc0 = __builtin_amdgcn_wmma_f32_16x16x32_f16(
            false, a.v, false, b0.v, (short)0, acc0, false, false);
        acc1 = __builtin_amdgcn_wmma_f32_16x16x32_f16(
            false, a.v, false, b1.v, (short)0, acc1, false, false);
    }

    int rbase = m0 + (hi ? 8 : 0);
    float bv0 = bias[cc0];
    float bv1 = bias[cc1];

    if constexpr (OUT16) {
        _Float16* C = (_Float16*)Cout;
        if (nc0 < Nstore) {
            #pragma unroll
            for (int r = 0; r < 8; ++r) {
                float v = acc0[r] + bv0;
                if (relu) v = fmaxf(v, 0.0f);
                if (nc0 >= N) v = 0.0f;     // write zeros into K-pad columns
                C[(size_t)(rbase + r) * Cstride + nc0] = (_Float16)v;
            }
        }
        if (nc1 < Nstore) {
            #pragma unroll
            for (int r = 0; r < 8; ++r) {
                float v = acc1[r] + bv1;
                if (relu) v = fmaxf(v, 0.0f);
                if (nc1 >= N) v = 0.0f;
                C[(size_t)(rbase + r) * Cstride + nc1] = (_Float16)v;
            }
        }
    } else {
        float* C = (float*)Cout;
        if (nc0 < Nstore) {
            #pragma unroll
            for (int r = 0; r < 8; ++r) {
                float v = acc0[r] + bv0;
                if (relu) v = fmaxf(v, 0.0f);
                C[(size_t)(rbase + r) * Cstride + nc0] = v;
            }
        }
        if (nc1 < Nstore) {
            #pragma unroll
            for (int r = 0; r < 8; ++r) {
                float v = acc1[r] + bv1;
                if (relu) v = fmaxf(v, 0.0f);
                C[(size_t)(rbase + r) * Cstride + nc1] = v;
            }
        }
    }
}

// ---------------------------------------------------------------------------
// Per-column sum / sumsq partial reduction (for batch-norm batch stats)
// ---------------------------------------------------------------------------
#define STAT_ROWS 128
__global__ void k_stats(const float* __restrict__ h2,
                        float* __restrict__ stats,   // [2*EMB]: sum, sumsq
                        int M) {
    int c = threadIdx.x;
    if (c >= EMB) return;
    int r0   = blockIdx.x * STAT_ROWS;
    int rend = min(r0 + STAT_ROWS, M);
    float s = 0.0f, q = 0.0f;
    for (int r = r0; r < rend; ++r) {
        float v = h2[(size_t)r * EMB + c];
        s += v;
        q += v * v;
    }
    atomic_add_f32(&stats[c], s);
    atomic_add_f32(&stats[EMB + c], q);
}

// ---------------------------------------------------------------------------
// BatchNorm (training-mode batch stats) + optional ReLU
// ---------------------------------------------------------------------------
__global__ void k_bn(const float* __restrict__ h2,
                     const float* __restrict__ stats,
                     const float* __restrict__ gamma,
                     const float* __restrict__ beta,
                     float* __restrict__ out, int relu) {
    int n = blockIdx.x;
    int c = threadIdx.x;
    if (c >= EMB) return;
    const float invN = 1.0f / (float)NNODES;
    float mu  = stats[c] * invN;
    float var = stats[EMB + c] * invN - mu * mu;
    float v = (h2[(size_t)n * EMB + c] - mu) * rsqrtf(var + EPS_BN) * gamma[c] + beta[c];
    if (relu) v = fmaxf(v, 0.0f);
    out[(size_t)n * EMB + c] = v;
}

// ---------------------------------------------------------------------------
extern "C" void kernel_launch(void* const* d_in, const int* in_sizes, int n_in,
                              void* d_out, int out_size, void* d_ws, size_t ws_size,
                              hipStream_t stream) {
    const int*   x          = (const int*)  d_in[0];
    const int*   edge_index = (const int*)  d_in[1];
    const int*   edge_attr  = (const int*)  d_in[2];
    const float* x_emb1     = (const float*)d_in[3];
    const float* x_emb2     = (const float*)d_in[4];
    const float* edge_emb1  = (const float*)d_in[5];   // [L,6,EMB]
    const float* edge_emb2  = (const float*)d_in[6];   // [L,3,EMB]
    const float* W1         = (const float*)d_in[7];   // [L,EMB,HID]
    const float* b1         = (const float*)d_in[8];   // [L,HID]
    const float* W2         = (const float*)d_in[9];   // [L,HID,EMB]
    const float* b2         = (const float*)d_in[10];  // [L,EMB]
    const float* gamma      = (const float*)d_in[11];  // [L,EMB]
    const float* beta       = (const float*)d_in[12];  // [L,EMB]
    float* out = (float*)d_out;

    // ---- workspace carve-up (all 16B aligned) ----
    float*    h     = (float*)d_ws;                          // 15.0 M f32
    float*    agg   = h     + (size_t)NNODES * EMB;          // 15.0 M f32 (reused as h2)
    float*    stats = agg   + (size_t)NNODES * EMB;          // 600 f32
    _Float16* aggH  = (_Float16*)(stats + 600);              // 16.0 M f16
    _Float16* hm    = aggH  + (size_t)NNODES * KP1;          // 30.4 M f16
    _Float16* w1t   = hm    + (size_t)NNODES * KP2;          // 600*320 f16
    _Float16* w2t   = w1t   + (size_t)HID * KP1;             // 300*608 f16

    k_node_emb<<<NNODES, 320, 0, stream>>>(x, x_emb1, x_emb2, h);

    const int tilesM = NNODES / 16;                          // 3125 (exact)

    for (int l = 0; l < NLAYER; ++l) {
        int nAgg = NNODES * EMB;
        k_zero<<<(nAgg + 255) / 256, 256, 0, stream>>>(agg, nAgg);
        k_zero<<<(2 * EMB + 255) / 256, 256, 0, stream>>>(stats, 2 * EMB);

        // transpose+convert this layer's weights to f16 [N, Kp] with K zero-pad
        k_cvt_w<<<HID, KP1, 0, stream>>>(W1 + (size_t)l * EMB * HID, w1t, EMB, HID, KP1);
        k_cvt_w<<<EMB, KP2, 0, stream>>>(W2 + (size_t)l * HID * EMB, w2t, HID, EMB, KP2);

        k_scatter<<<NEDGES + NNODES, 320, 0, stream>>>(
            h, edge_index, edge_attr,
            edge_emb1 + (size_t)l * 6 * EMB,
            edge_emb2 + (size_t)l * 3 * EMB, agg);

        k_cvt_agg<<<NNODES, KP1, 0, stream>>>(agg, aggH);

        // GEMM1: hm(f16,[M,KP2]) = relu(aggH @ W1 + b1)   [50000,320]x[320,600]
        {
            int tilesN2 = (HID + 31) / 32;                   // 19
            int waves   = tilesM * tilesN2;
            int blocks  = (waves + 3) / 4;                   // 4 waves / 128-thread block
            k_gemm_wmma_f16<KP1, true><<<blocks, 128, 0, stream>>>(
                aggH, w1t, b1 + (size_t)l * HID, hm,
                HID, KP2, KP2, tilesM, 1);
        }

        // GEMM2: h2(f32,[M,EMB]) = hm @ W2 + b2           [50000,608]x[608,300]
        float* h2 = agg;  // reuse (agg dead after cvt)
        {
            int tilesN2 = (EMB + 31) / 32;                   // 10
            int waves   = tilesM * tilesN2;
            int blocks  = (waves + 3) / 4;
            k_gemm_wmma_f16<KP2, false><<<blocks, 128, 0, stream>>>(
                hm, w2t, b2 + (size_t)l * EMB, h2,
                EMB, EMB, EMB, tilesM, 0);
        }

        k_stats<<<(NNODES + STAT_ROWS - 1) / STAT_ROWS, 320, 0, stream>>>(h2, stats, NNODES);

        float* dst = (l == NLAYER - 1) ? out : h;
        k_bn<<<NNODES, 320, 0, stream>>>(
            h2, stats, gamma + (size_t)l * EMB, beta + (size_t)l * EMB,
            dst, (l < NLAYER - 1) ? 1 : 0);
    }
}
